// DecompLayer_63952063037875
// MI455X (gfx1250) — matrix operand order repaired and checked by reading, not compiled, not run on hardware
//
#include <hip/hip_runtime.h>

// ---- problem constants (match reference) ----
#define Bn   32
#define Cc   256
#define Hh   96
#define Ww   96
#define HW   (Hh*Ww)      // 9216
#define Rr   128
#define Ff   512
#define OH   94
#define OW   94
#define OHW  (OH*OW)      // 8836

// ---- GEMM tiling ----
#define KT   32           // k-step per stage
#define MT   128          // output rows per block (weight dim)
#define NT   128          // output cols per block (spatial dim)
#define LDA  40           // padded LDS row stride in bf16 elems (80B, 16B aligned)

typedef __bf16 bf16_t;
typedef __attribute__((ext_vector_type(16))) __bf16 v16bf;
typedef __attribute__((ext_vector_type(8)))  __bf16 v8bf;
typedef __attribute__((ext_vector_type(4)))  __bf16 v4bf;
typedef __attribute__((ext_vector_type(8)))  float  v8f;

union Frag16 { v16bf v; v8bf h[2]; };

// =====================================================================
// Kernel 1: y1[b, r, m] = sum_c f3[c, r] * x[b, c, m]    (m = h*W+w)
// WMMA A = f3^T (M=r, K=c), B = x[b] (K=c, N=m). D stored bf16, coalesced in m.
// Double-buffered LDS pipeline over 8 K-steps of 32.
// =====================================================================
__global__ __launch_bounds__(256) void gemm_cr_kernel(
    const float* __restrict__ x, const float* __restrict__ f3,
    bf16_t* __restrict__ y1)
{
  __shared__ __align__(16) bf16_t Al[2][MT * LDA];
  __shared__ __align__(16) bf16_t Bl[2][NT * LDA];

  const int tid   = threadIdx.x;
  const int wave  = tid >> 5;
  const int lane  = tid & 31;
  const int half  = lane >> 4;
  const int lan16 = lane & 15;
  const int b     = blockIdx.y;
  const int m0    = blockIdx.x * NT;

  const float* xb = x + (size_t)b * Cc * HW;

  // Thread-fixed staging coordinates: rows k = wave + it*8, cols = lane*4.
  const int kr  = wave;        // base k row for this thread
  const int c4  = lane * 4;    // contiguous column (A: r, B: spatial m)

  v8f acc[8] = {};
  float4 ra[4], rb[4];

  constexpr int STEPS = Cc / KT;   // 8

  // Prologue: load step 0, spill to buffer 0.
  #pragma unroll
  for (int it = 0; it < 4; ++it) {
    int k = kr + it * 8;
    ra[it] = *(const float4*)&f3[(size_t)k * Rr + c4];
    rb[it] = *(const float4*)&xb[(size_t)k * HW + m0 + c4];
  }
  #pragma unroll
  for (int it = 0; it < 4; ++it) {
    int k = kr + it * 8;
    const float* fa = (const float*)&ra[it];
    const float* fb = (const float*)&rb[it];
    #pragma unroll
    for (int j = 0; j < 4; ++j) {
      Al[0][(c4 + j) * LDA + k] = (bf16_t)fa[j];
      Bl[0][(c4 + j) * LDA + k] = (bf16_t)fb[j];
    }
  }

  for (int s = 0; s < STEPS; ++s) {
    const int cur = s & 1;
    __syncthreads();

    const bool has_next = (s + 1 < STEPS);
    if (has_next) {
      int k0n = (s + 1) * KT;
      #pragma unroll
      for (int it = 0; it < 4; ++it) {
        int k = k0n + kr + it * 8;
        ra[it] = *(const float4*)&f3[(size_t)k * Rr + c4];
        rb[it] = *(const float4*)&xb[(size_t)k * HW + m0 + c4];
      }
      if (s + 2 < STEPS)
        __builtin_prefetch(&xb[(size_t)((s + 2) * KT + kr) * HW + m0 + c4], 0, 1);
    }

    // ---- compute from buffer `cur` ----
    Frag16 a;
    const bf16_t* pa = &Al[cur][(wave * 16 + lan16) * LDA];
    a.h[0] = *(const v8bf*)(pa + 16 * half);
    a.h[1] = *(const v8bf*)(pa + 16 * half + 8);

    const bf16_t* Bbase = &Bl[cur][lan16 * LDA + 16 * half];
    Frag16 fbr[2];
    fbr[0].h[0] = *(const v8bf*)(Bbase);
    fbr[0].h[1] = *(const v8bf*)(Bbase + 8);
    #pragma unroll
    for (int nt = 0; nt < 8; ++nt) {
      if (nt < 7) {
        const bf16_t* pb = Bbase + (nt + 1) * 16 * LDA;
        fbr[(nt + 1) & 1].h[0] = *(const v8bf*)(pb);
        fbr[(nt + 1) & 1].h[1] = *(const v8bf*)(pb + 8);
      }
      acc[nt] = __builtin_amdgcn_wmma_f32_16x16x32_bf16(
          false, a.v, false, fbr[nt & 1].v, (short)0, acc[nt], false, false);
    }

    // ---- spill next tile to the other buffer ----
    if (has_next) {
      #pragma unroll
      for (int it = 0; it < 4; ++it) {
        int k = kr + it * 8;
        const float* fa = (const float*)&ra[it];
        const float* fb = (const float*)&rb[it];
        #pragma unroll
        for (int j = 0; j < 4; ++j) {
          Al[cur ^ 1][(c4 + j) * LDA + k] = (bf16_t)fa[j];
          Bl[cur ^ 1][(c4 + j) * LDA + k] = (bf16_t)fb[j];
        }
      }
    }
  }

  // Store D: M = r (row-major), N = spatial (contiguous lanes -> coalesced).
  bf16_t* ob = y1 + (size_t)b * Rr * HW;
  #pragma unroll
  for (int nt = 0; nt < 8; ++nt) {
    #pragma unroll
    for (int rr = 0; rr < 8; ++rr) {
      int Mrow = wave * 16 + rr + 8 * half;
      int Ncol = m0 + nt * 16 + lan16;
      ob[(size_t)Mrow * HW + Ncol] = (bf16_t)acc[nt][rr];
    }
  }
}

// =====================================================================
// Kernel 2: separable 3x3 depthwise (rank-1 taps f1[h,r]*f2[w,r]), VALID.
// =====================================================================
__global__ __launch_bounds__(128) void sepconv3x3_kernel(
    const bf16_t* __restrict__ y1, const float* __restrict__ f1,
    const float* __restrict__ f2, bf16_t* __restrict__ y3)
{
  const int j = threadIdx.x;
  const int i = blockIdx.x;
  const int r = blockIdx.y;
  const int b = blockIdx.z;
  if (j >= OW) return;

  const float t10 = f1[0 * Rr + r], t11 = f1[1 * Rr + r], t12 = f1[2 * Rr + r];
  const float t20 = f2[0 * Rr + r], t21 = f2[1 * Rr + r], t22 = f2[2 * Rr + r];

  const bf16_t* src = y1 + (size_t)(b * Rr + r) * HW + i * Ww + j;
  float acc = 0.f;
  #pragma unroll
  for (int h = 0; h < 3; ++h) {
    const float th = (h == 0) ? t10 : ((h == 1) ? t11 : t12);
    const bf16_t* row = src + h * Ww;
    float s = (float)row[0] * t20 + (float)row[1] * t21 + (float)row[2] * t22;
    acc += th * s;
  }
  y3[(size_t)(b * Rr + r) * OHW + i * OW + j] = (bf16_t)acc;
}

// =====================================================================
// Kernel 3: out[b, f, m] = sum_r f0[f, r] * y3[b, r, m]
// WMMA A = f0 (M=f, K=r), B = y3[b] (K=r, N=m). 4 K-steps of 32.
// =====================================================================
__global__ __launch_bounds__(256) void gemm_rf_kernel(
    const bf16_t* __restrict__ y3, const float* __restrict__ f0,
    float* __restrict__ out)
{
  __shared__ __align__(16) bf16_t Al[2][MT * LDA];
  __shared__ __align__(16) bf16_t Bl[2][NT * LDA];

  const int tid   = threadIdx.x;
  const int wave  = tid >> 5;
  const int lane  = tid & 31;
  const int half  = lane >> 4;
  const int lan16 = lane & 15;
  const int b     = blockIdx.z;
  const int fbase = blockIdx.y * MT;
  const int m0    = blockIdx.x * NT;

  const bf16_t* yb = y3 + (size_t)b * Rr * OHW;

  // A staging: m = itA*32 + (tid>>3), 4 contiguous k at k4 = (tid&7)*4.
  const int am  = tid >> 3;
  const int ak4 = (tid & 7) * 4;
  // B staging: k = itB*16 + (tid>>4), 8 contiguous n at n8 = (tid&15)*8 (clamped).
  const int bk  = tid >> 4;
  const int bn8 = (tid & 15) * 8;
  const int bcol = (m0 + bn8 <= OHW - 8) ? (m0 + bn8) : (OHW - 8);

  v8f acc[8] = {};
  float4 ra[4];
  v4bf  rb[2][2];

  constexpr int STEPS = Rr / KT;   // 4

  // Prologue: load step 0, spill to buffer 0.
  #pragma unroll
  for (int it = 0; it < 4; ++it)
    ra[it] = *(const float4*)&f0[(size_t)(fbase + it * 32 + am) * Rr + ak4];
  #pragma unroll
  for (int it = 0; it < 2; ++it) {
    const bf16_t* p = &yb[(size_t)(it * 16 + bk) * OHW + bcol];
    rb[it][0] = *(const v4bf*)(p);
    rb[it][1] = *(const v4bf*)(p + 4);
  }
  #pragma unroll
  for (int it = 0; it < 4; ++it) {
    const float* fa = (const float*)&ra[it];
    v4bf cv = { (bf16_t)fa[0], (bf16_t)fa[1], (bf16_t)fa[2], (bf16_t)fa[3] };
    *(v4bf*)&Al[0][(it * 32 + am) * LDA + ak4] = cv;   // ds_store_b64
  }
  #pragma unroll
  for (int it = 0; it < 2; ++it) {
    int k = it * 16 + bk;
    #pragma unroll
    for (int j = 0; j < 8; ++j)
      Bl[0][(bn8 + j) * LDA + k] = ((const bf16_t*)&rb[it])[j];
  }

  for (int s = 0; s < STEPS; ++s) {
    const int cur = s & 1;
    __syncthreads();

    const bool has_next = (s + 1 < STEPS);
    if (has_next) {
      int k0n = (s + 1) * KT;
      #pragma unroll
      for (int it = 0; it < 4; ++it)
        ra[it] = *(const float4*)&f0[(size_t)(fbase + it * 32 + am) * Rr + k0n + ak4];
      #pragma unroll
      for (int it = 0; it < 2; ++it) {
        const bf16_t* p = &yb[(size_t)(k0n + it * 16 + bk) * OHW + bcol];
        rb[it][0] = *(const v4bf*)(p);
        rb[it][1] = *(const v4bf*)(p + 4);
      }
      if (s + 2 < STEPS)
        __builtin_prefetch(&yb[(size_t)((s + 2) * KT + bk) * OHW + bcol], 0, 1);
    }

    // ---- compute from buffer `cur` ----
    Frag16 a;
    const bf16_t* pa = &Al[cur][(wave * 16 + lan16) * LDA];
    a.h[0] = *(const v8bf*)(pa + 16 * half);
    a.h[1] = *(const v8bf*)(pa + 16 * half + 8);

    const bf16_t* Bbase = &Bl[cur][lan16 * LDA + 16 * half];
    Frag16 fbr[2];
    fbr[0].h[0] = *(const v8bf*)(Bbase);
    fbr[0].h[1] = *(const v8bf*)(Bbase + 8);
    #pragma unroll
    for (int nt = 0; nt < 8; ++nt) {
      if (nt < 7) {
        const bf16_t* pb = Bbase + (nt + 1) * 16 * LDA;
        fbr[(nt + 1) & 1].h[0] = *(const v8bf*)(pb);
        fbr[(nt + 1) & 1].h[1] = *(const v8bf*)(pb + 8);
      }
      acc[nt] = __builtin_amdgcn_wmma_f32_16x16x32_bf16(
          false, a.v, false, fbr[nt & 1].v, (short)0, acc[nt], false, false);
    }

    // ---- spill next tile to the other buffer ----
    if (has_next) {
      #pragma unroll
      for (int it = 0; it < 4; ++it) {
        const float* fa = (const float*)&ra[it];
        v4bf cv = { (bf16_t)fa[0], (bf16_t)fa[1], (bf16_t)fa[2], (bf16_t)fa[3] };
        *(v4bf*)&Al[cur ^ 1][(it * 32 + am) * LDA + ak4] = cv;
      }
      #pragma unroll
      for (int it = 0; it < 2; ++it) {
        int k = it * 16 + bk;
        #pragma unroll
        for (int j = 0; j < 8; ++j)
          Bl[cur ^ 1][(bn8 + j) * LDA + k] = ((const bf16_t*)&rb[it])[j];
      }
    }
  }

  // Store fp32 output, coalesced along spatial axis (lanes = N).
  float* ob = out + (size_t)b * Ff * OHW;
  #pragma unroll
  for (int nt = 0; nt < 8; ++nt) {
    #pragma unroll
    for (int rr = 0; rr < 8; ++rr) {
      int Mrow = fbase + wave * 16 + rr + 8 * half;
      int Ncol = m0 + nt * 16 + lan16;
      if (Ncol < OHW)
        ob[(size_t)Mrow * OHW + Ncol] = acc[nt][rr];
    }
  }
}

// =====================================================================
extern "C" void kernel_launch(void* const* d_in, const int* in_sizes, int n_in,
                              void* d_out, int out_size, void* d_ws, size_t ws_size,
                              hipStream_t stream) {
  (void)in_sizes; (void)n_in; (void)out_size; (void)ws_size;
  const float* x  = (const float*)d_in[0];   // [B,C,H,W]
  const float* f0 = (const float*)d_in[1];   // [F,R]
  const float* f1 = (const float*)d_in[2];   // [FH,R]
  const float* f2 = (const float*)d_in[3];   // [FW,R]
  const float* f3 = (const float*)d_in[4];   // [C,R]
  float* out = (float*)d_out;                // [B,F,OH,OW]

  bf16_t* y1 = (bf16_t*)d_ws;                                    // [B,R,HW]  bf16
  bf16_t* y3 = (bf16_t*)((char*)d_ws +
                         (size_t)Bn * Rr * HW * sizeof(bf16_t)); // [B,R,OHW] bf16

  gemm_cr_kernel<<<dim3(HW / NT, Bn), 256, 0, stream>>>(x, f3, y1);
  sepconv3x3_kernel<<<dim3(OH, Rr, Bn), 128, 0, stream>>>(y1, f1, f2, y3);
  gemm_rf_kernel<<<dim3((OHW + NT - 1) / NT, Ff / MT, Bn), 256, 0, stream>>>(
      y3, f0, out);
}
